// dyMEANModel_63024350101679
// MI455X (gfx1250) — matrix operand order repaired and checked by reading, not compiled, not run on hardware
//
#include <hip/hip_runtime.h>

// ---------------------------------------------------------------------------
// Model constants (match reference)
// ---------------------------------------------------------------------------
#define NN     6144     // nodes
#define SEG    512
#define NCH    14       // channels
#define ED     64       // embed dim
#define HDIM   128      // hidden dim
#define NCLS   25
#define KNB    9        // neighbors
#define NL     3        // layers
#define NROUND 3

typedef __attribute__((ext_vector_type(16))) _Float16 v16h;
typedef __attribute__((ext_vector_type(8)))  float    v8f;

// Fast silu: x * rcp(1 + exp(-x)).  v_rcp_f32 (~1 ulp) instead of the IEEE
// division sequence; results feed f16 WMMA operands so this is ample.
__device__ __forceinline__ float silu_f(float x) {
  return x * __builtin_amdgcn_rcpf(1.0f + __expf(-x));
}

template<int NT>
__device__ __forceinline__ void vzero(v8f (&acc)[NT]) {
#pragma unroll
  for (int t = 0; t < NT; ++t) {
#pragma unroll
    for (int r = 0; r < 8; ++r) acc[t][r] = 0.0f;
  }
}

// GEMM with A in LDS ([16][lda] f16, rows = edge/node within tile) and B in
// fragment-major f16 global memory (pre-swizzled). NT output 16x16 tiles,
// KS K-steps of 32. Accumulates into acc (f32).
template<int NT, int KS>
__device__ __forceinline__ void gemm_frag(const _Float16* __restrict__ A, int lda,
                                          const _Float16* __restrict__ Bf,
                                          int lane, v8f (&acc)[NT]) {
  const int row   = lane & 15;
  const int kaoff = (lane & 16) ? 8 : 0;
#pragma unroll
  for (int s = 0; s < KS; ++s) {
    v16h a;
    const _Float16* ap = A + row * lda + s * 32 + kaoff;
#pragma unroll
    for (int j = 0; j < 8; ++j) { a[j] = ap[j]; a[j + 8] = ap[16 + j]; }
#pragma unroll
    for (int t = 0; t < NT; ++t) {
      v16h b = *(const v16h*)(Bf + ((size_t)(s * NT + t) * 32 + lane) * 16);
      acc[t] = __builtin_amdgcn_wmma_f32_16x16x32_f16(
          false, a, false, b, (short)0, acc[t], false, false);
    }
  }
}

// ---------------------------------------------------------------------------
// Weight swizzle: f32 [krows x ncsrc] -> f16 fragment-major, padded to
// [Ktot x ncols] with zeros.  dst[((s*NT+t)*32+lane)*16+j] =
//   W[s*32 + (lane>=16?8:0) + j + (j>=8?8:0)][t*16 + (lane&15)]
// ---------------------------------------------------------------------------
__global__ void swz_kernel(_Float16* __restrict__ dst, const float* __restrict__ src,
                           int Ktot, int ncols, int krows, int ncsrc) {
  int total = (Ktot >> 5) * (ncols >> 4) * 512;
  int i = blockIdx.x * 256 + threadIdx.x;
  if (i >= total) return;
  int j    = i & 15;
  int lane = (i >> 4) & 31;
  int ft   = i >> 9;
  int NT   = ncols >> 4;
  int t    = ft % NT;
  int s    = ft / NT;
  int K    = s * 32 + ((lane & 16) ? 8 : 0) + j + ((j >= 8) ? 8 : 0);
  int col  = t * 16 + (lane & 15);
  float v = (K < krows && col < ncsrc) ? src[(size_t)K * ncsrc + col] : 0.0f;
  dst[i] = (_Float16)v;
}

__global__ void copy_f32_kernel(float* __restrict__ dst, const float* __restrict__ src, int n) {
  int i = blockIdx.x * 256 + threadIdx.x;
  if (i < n) dst[i] = src[i];
}

// ---------------------------------------------------------------------------
// KNN: one block per segment of 512 nodes; top-9 nearest by channel-1 coords.
// ---------------------------------------------------------------------------
__global__ __launch_bounds__(512) void knn_kernel(const float* __restrict__ X,
                                                  int* __restrict__ nbr) {
  __shared__ float ca[SEG][3];
  const int tid = threadIdx.x;
  const int seg = blockIdx.x;
  const int gi  = seg * SEG + tid;
  ca[tid][0] = X[gi * (NCH * 3) + 3 + 0];
  ca[tid][1] = X[gi * (NCH * 3) + 3 + 1];
  ca[tid][2] = X[gi * (NCH * 3) + 3 + 2];
  __syncthreads();
  const float px = ca[tid][0], py = ca[tid][1], pz = ca[tid][2];
  float bd[KNB];
  int   bj[KNB];
#pragma unroll
  for (int k = 0; k < KNB; ++k) { bd[k] = 3.0e38f; bj[k] = 0; }
  for (int j = 0; j < SEG; ++j) {
    if (j == tid) continue;
    float dx = px - ca[j][0], dy = py - ca[j][1], dz = pz - ca[j][2];
    float d = dx * dx + dy * dy + dz * dz;
    int jj = j;
#pragma unroll
    for (int p = 0; p < KNB; ++p) {           // stable sorted insert
      bool sw = d < bd[p];
      float td = sw ? bd[p] : d;  int tj = sw ? bj[p] : jj;
      bd[p] = sw ? d : bd[p];     bj[p] = sw ? jj : bj[p];
      d = td; jj = tj;
    }
  }
#pragma unroll
  for (int k = 0; k < KNB; ++k) nbr[gi * KNB + k] = seg * SEG + bj[k];
}

// ---------------------------------------------------------------------------
// Embed + memory MLP + input projection:
//   H0 = embed[S] (+ silu(silu(H)@mW1+mb1)@mW2+mb2 if use_mem)
//   H  = H0 @ W_in + b_in
// 4 waves/block, 16 nodes/wave. In-place on H (each wave touches own rows).
// ---------------------------------------------------------------------------
__global__ __launch_bounds__(128) void embed_kernel(
    float* __restrict__ H, const int* __restrict__ S, const float* __restrict__ embed,
    const _Float16* __restrict__ W_inf, const float* __restrict__ b_in,
    const _Float16* __restrict__ mW1f, const float* __restrict__ mb1,
    const _Float16* __restrict__ mW2f, const float* __restrict__ mb2, int use_mem) {
  extern __shared__ char smem[];
  const int tid  = threadIdx.x;
  const int wave = tid >> 5;
  const int lane = tid & 31;
  _Float16* act = (_Float16*)smem + wave * (16 * HDIM);
  _Float16* H0  = (_Float16*)smem + 4 * (16 * HDIM) + wave * (16 * ED);
  const int g    = blockIdx.x * 4 + wave;
  const int row  = lane & 15;
  const int hi8  = (lane & 16) ? 8 : 0;

  if (use_mem) {
    v8f acc[8]; vzero(acc);
#pragma unroll
    for (int s = 0; s < 4; ++s) {          // K = 128, A = silu(H)
      v16h a;
#pragma unroll
      for (int j = 0; j < 16; ++j) {
        int K = s * 32 + hi8 + j + ((j >= 8) ? 8 : 0);
        a[j] = (_Float16)silu_f(H[(size_t)(g * 16 + row) * HDIM + K]);
      }
#pragma unroll
      for (int t = 0; t < 8; ++t) {
        v16h b = *(const v16h*)(mW1f + ((size_t)(s * 8 + t) * 32 + lane) * 16);
        acc[t] = __builtin_amdgcn_wmma_f32_16x16x32_f16(false, a, false, b, (short)0,
                                                        acc[t], false, false);
      }
    }
#pragma unroll
    for (int t = 0; t < 8; ++t) {
      int col = t * 16 + row; float bias = mb1[col];
#pragma unroll
      for (int r = 0; r < 8; ++r)
        act[(r + hi8) * HDIM + col] = (_Float16)silu_f(acc[t][r] + bias);
    }
    __syncthreads();
    v8f acc2[4]; vzero(acc2);
    gemm_frag<4, 4>(act, HDIM, mW2f, lane, acc2);      // [16x128]@[128x64]
#pragma unroll
    for (int t = 0; t < 4; ++t) {
      int col = t * 16 + row; float bias = mb2[col];
#pragma unroll
      for (int r = 0; r < 8; ++r) {
        int rr = r + hi8; int node = g * 16 + rr;
        H0[rr * ED + col] =
            (_Float16)(acc2[t][r] + bias + embed[(size_t)S[node] * ED + col]);
      }
    }
  } else {
    for (int idx = lane; idx < 16 * ED; idx += 32) {
      int rr = idx >> 6, col = idx & 63;
      H0[idx] = (_Float16)embed[(size_t)S[g * 16 + rr] * ED + col];
    }
  }
  __syncthreads();
  v8f acc3[8]; vzero(acc3);
  gemm_frag<8, 2>(H0, ED, W_inf, lane, acc3);          // [16x64]@[64x128]
#pragma unroll
  for (int t = 0; t < 8; ++t) {
    int col = t * 16 + row; float bias = b_in[col];
#pragma unroll
    for (int r = 0; r < 8; ++r)
      H[(size_t)(g * 16 + r + hi8) * HDIM + col] = acc3[t][r] + bias;
  }
}

// ---------------------------------------------------------------------------
// Edge kernel: one block = 16 nodes = 144 edges = 9 tiles of 16 edges
// (one per wave). ein(257) -> eW1 -> silu -> eW2 -> silu = msg
// scale = silu(msg@cW1+cb1)@cW2+cb2 ; X_next = X + mean_k(diff*scale)
// agg = sum_k msg
// ---------------------------------------------------------------------------
__global__ __launch_bounds__(288) void edge_kernel(
    const float* __restrict__ Xc, float* __restrict__ Xn,
    const float* __restrict__ H, float* __restrict__ agg,
    const int* __restrict__ nbr, const float* __restrict__ channel_w,
    const _Float16* __restrict__ eW1f, const float* __restrict__ eb1,
    const _Float16* __restrict__ eW2f, const float* __restrict__ eb2,
    const _Float16* __restrict__ cW1f, const float* __restrict__ cb1,
    const float* __restrict__ cW2l, const float* __restrict__ cb2l) {
  extern __shared__ char smem[];
  float*     diffb  = (float*)smem;                        // [144][42]
  float*     scaleb = diffb + 144 * 42;                    // [144]
  float*     msgb   = scaleb + 144;                        // [144][128]
  _Float16*  actb   = (_Float16*)(msgb + 144 * HDIM);      // [9][16][128]

  const int tid  = threadIdx.x;
  const int wave = tid >> 5;
  const int lane = tid & 31;
  const int g    = blockIdx.x;
  const int row  = lane & 15;
  const int hi8  = (lane & 16) ? 8 : 0;
  _Float16* act = actb + wave * (16 * HDIM);

  // --- phase 1: diff into LDS ------------------------------------------------
  for (int idx = tid; idx < 144 * 42; idx += 288) {
    int e = idx / 42, d = idx % 42;
    int eg = g * 144 + e;
    int i = eg / KNB;
    int j = nbr[eg];
    diffb[idx] = Xc[(size_t)i * 42 + d] - Xc[(size_t)j * 42 + d];
  }
  __syncthreads();

  // per-lane row info + radial (redundant across the two half-waves)
  const int eg_row = g * 144 + wave * 16 + row;
  const int i_row  = eg_row / KNB;
  const int j_row  = nbr[eg_row];
  float radial = 0.0f;
  {
    const float* dp = diffb + (wave * 16 + row) * 42;
#pragma unroll
    for (int c = 0; c < NCH; ++c) {
      float dx = dp[3 * c], dy = dp[3 * c + 1], dz = dp[3 * c + 2];
      radial += channel_w[c] * (dx * dx + dy * dy + dz * dz);
    }
  }

  // --- GEMM1: ein @ eW1  (K padded 257 -> 288, 9 steps, 8 tiles) -------------
  v8f acc[8]; vzero(acc);
#pragma unroll
  for (int s = 0; s < 9; ++s) {
    v16h a;
#pragma unroll
    for (int j = 0; j < 16; ++j) {
      int K = s * 32 + hi8 + j + ((j >= 8) ? 8 : 0);
      float v;
      if (K < HDIM)            v = H[(size_t)i_row * HDIM + K];
      else if (K < 2 * HDIM)   v = H[(size_t)j_row * HDIM + (K - HDIM)];
      else                     v = (K == 2 * HDIM) ? radial : 0.0f;
      a[j] = (_Float16)v;
    }
#pragma unroll
    for (int t = 0; t < 8; ++t) {
      v16h b = *(const v16h*)(eW1f + ((size_t)(s * 8 + t) * 32 + lane) * 16);
      acc[t] = __builtin_amdgcn_wmma_f32_16x16x32_f16(false, a, false, b, (short)0,
                                                      acc[t], false, false);
    }
  }
#pragma unroll
  for (int t = 0; t < 8; ++t) {
    int col = t * 16 + row; float bias = eb1[col];
#pragma unroll
    for (int r = 0; r < 8; ++r)
      act[(r + hi8) * HDIM + col] = (_Float16)silu_f(acc[t][r] + bias);
  }
  __syncthreads();

  // --- GEMM2: @ eW2 -> msg ---------------------------------------------------
  v8f acc2[8]; vzero(acc2);
  gemm_frag<8, 4>(act, HDIM, eW2f, lane, acc2);
#pragma unroll
  for (int t = 0; t < 8; ++t) {
    int col = t * 16 + row; float bias = eb2[col];
#pragma unroll
    for (int r = 0; r < 8; ++r) {
      int rr = r + hi8;
      float m = silu_f(acc2[t][r] + bias);
      msgb[(wave * 16 + rr) * HDIM + col] = m;     // for agg reduction
      act[rr * HDIM + col] = (_Float16)m;          // A for GEMM3
    }
  }
  __syncthreads();

  // --- GEMM3: @ cW1, silu ----------------------------------------------------
  v8f acc3[8]; vzero(acc3);
  gemm_frag<8, 4>(act, HDIM, cW1f, lane, acc3);
#pragma unroll
  for (int t = 0; t < 8; ++t) {
    int col = t * 16 + row; float bias = cb1[col];
#pragma unroll
    for (int r = 0; r < 8; ++r) acc3[t][r] = silu_f(acc3[t][r] + bias);
  }
  // --- GEMM4: 128 -> 1 dot with cW2 + cross-lane reduce ----------------------
  float p[8];
#pragma unroll
  for (int r = 0; r < 8; ++r) p[r] = 0.0f;
#pragma unroll
  for (int t = 0; t < 8; ++t) {
    float w = cW2l[t * 16 + row];
#pragma unroll
    for (int r = 0; r < 8; ++r) p[r] += acc3[t][r] * w;
  }
#pragma unroll
  for (int m = 1; m < 16; m <<= 1) {
#pragma unroll
    for (int r = 0; r < 8; ++r) p[r] += __shfl_xor(p[r], m, 32);
  }
  if (row == 0) {
    float cb = cb2l[0];
#pragma unroll
    for (int r = 0; r < 8; ++r) scaleb[wave * 16 + r + hi8] = p[r] + cb;
  }
  __syncthreads();

  // --- reductions: agg (sum_k msg) and X update ------------------------------
  for (int idx = tid; idx < 16 * HDIM; idx += 288) {
    int nl = idx >> 7, col = idx & 127;
    float s = 0.0f;
#pragma unroll
    for (int k = 0; k < KNB; ++k) s += msgb[(nl * KNB + k) * HDIM + col];
    agg[(size_t)(g * 16 + nl) * HDIM + col] = s;
  }
  for (int idx = tid; idx < 16 * 42; idx += 288) {
    int nl = idx / 42, d = idx % 42;
    float s = 0.0f;
#pragma unroll
    for (int k = 0; k < KNB; ++k)
      s += diffb[(nl * KNB + k) * 42 + d] * scaleb[nl * KNB + k];
    size_t gi = (size_t)(g * 16 + nl) * 42 + d;
    Xn[gi] = Xc[gi] + s * (1.0f / 9.0f);
  }
}

// ---------------------------------------------------------------------------
// Node update: H += silu(cat(H, agg) @ nW1 + nb1) @ nW2 + nb2
// ---------------------------------------------------------------------------
__global__ __launch_bounds__(128) void node_kernel(
    float* __restrict__ H, const float* __restrict__ agg,
    const _Float16* __restrict__ nW1f, const float* __restrict__ nb1,
    const _Float16* __restrict__ nW2f, const float* __restrict__ nb2) {
  extern __shared__ char smem[];
  const int tid  = threadIdx.x;
  const int wave = tid >> 5;
  const int lane = tid & 31;
  _Float16* act = (_Float16*)smem + wave * (16 * HDIM);
  const int g   = blockIdx.x * 4 + wave;
  const int row = lane & 15;
  const int hi8 = (lane & 16) ? 8 : 0;

  v8f acc[8]; vzero(acc);
#pragma unroll
  for (int s = 0; s < 8; ++s) {          // K = 256 : [H | agg]
    v16h a;
#pragma unroll
    for (int j = 0; j < 16; ++j) {
      int K = s * 32 + hi8 + j + ((j >= 8) ? 8 : 0);
      float v = (K < HDIM) ? H[(size_t)(g * 16 + row) * HDIM + K]
                           : agg[(size_t)(g * 16 + row) * HDIM + (K - HDIM)];
      a[j] = (_Float16)v;
    }
#pragma unroll
    for (int t = 0; t < 8; ++t) {
      v16h b = *(const v16h*)(nW1f + ((size_t)(s * 8 + t) * 32 + lane) * 16);
      acc[t] = __builtin_amdgcn_wmma_f32_16x16x32_f16(false, a, false, b, (short)0,
                                                      acc[t], false, false);
    }
  }
#pragma unroll
  for (int t = 0; t < 8; ++t) {
    int col = t * 16 + row; float bias = nb1[col];
#pragma unroll
    for (int r = 0; r < 8; ++r)
      act[(r + hi8) * HDIM + col] = (_Float16)silu_f(acc[t][r] + bias);
  }
  __syncthreads();
  v8f acc2[8]; vzero(acc2);
  gemm_frag<8, 4>(act, HDIM, nW2f, lane, acc2);
#pragma unroll
  for (int t = 0; t < 8; ++t) {
    int col = t * 16 + row; float bias = nb2[col];
#pragma unroll
    for (int r = 0; r < 8; ++r)
      H[(size_t)(g * 16 + r + hi8) * HDIM + col] += acc2[t][r] + bias;
  }
}

// ---------------------------------------------------------------------------
// Output head: logits = silu(silu(H)@oW1+ob1)@oW2+ob2   (oW2 padded 25->32)
// ---------------------------------------------------------------------------
__global__ __launch_bounds__(128) void logits_kernel(
    const float* __restrict__ H,
    const _Float16* __restrict__ oW1f, const float* __restrict__ ob1,
    const _Float16* __restrict__ oW2f, const float* __restrict__ ob2,
    float* __restrict__ out) {
  extern __shared__ char smem[];
  const int tid  = threadIdx.x;
  const int wave = tid >> 5;
  const int lane = tid & 31;
  _Float16* act = (_Float16*)smem + wave * (16 * HDIM);
  const int g   = blockIdx.x * 4 + wave;
  const int row = lane & 15;
  const int hi8 = (lane & 16) ? 8 : 0;

  v8f acc[8]; vzero(acc);
#pragma unroll
  for (int s = 0; s < 4; ++s) {          // A = silu(H)
    v16h a;
#pragma unroll
    for (int j = 0; j < 16; ++j) {
      int K = s * 32 + hi8 + j + ((j >= 8) ? 8 : 0);
      a[j] = (_Float16)silu_f(H[(size_t)(g * 16 + row) * HDIM + K]);
    }
#pragma unroll
    for (int t = 0; t < 8; ++t) {
      v16h b = *(const v16h*)(oW1f + ((size_t)(s * 8 + t) * 32 + lane) * 16);
      acc[t] = __builtin_amdgcn_wmma_f32_16x16x32_f16(false, a, false, b, (short)0,
                                                      acc[t], false, false);
    }
  }
#pragma unroll
  for (int t = 0; t < 8; ++t) {
    int col = t * 16 + row; float bias = ob1[col];
#pragma unroll
    for (int r = 0; r < 8; ++r)
      act[(r + hi8) * HDIM + col] = (_Float16)silu_f(acc[t][r] + bias);
  }
  __syncthreads();
  v8f acc2[2]; vzero(acc2);
  gemm_frag<2, 4>(act, HDIM, oW2f, lane, acc2);
#pragma unroll
  for (int t = 0; t < 2; ++t) {
    int col = t * 16 + row;
    if (col < NCLS) {
      float bias = ob2[col];
#pragma unroll
      for (int r = 0; r < 8; ++r)
        out[(size_t)(g * 16 + r + hi8) * NCLS + col] = acc2[t][r] + bias;
    }
  }
}

// ---------------------------------------------------------------------------
// Host orchestration
// ---------------------------------------------------------------------------
extern "C" void kernel_launch(void* const* d_in, const int* in_sizes, int n_in,
                              void* d_out, int out_size, void* d_ws, size_t ws_size,
                              hipStream_t stream) {
  const float* X_in      = (const float*)d_in[0];
  const int*   S         = (const int*)d_in[1];
  const float* embed     = (const float*)d_in[3];
  const float* W_in      = (const float*)d_in[4];
  const float* b_in      = (const float*)d_in[5];
  const float* channel_w = (const float*)d_in[6];
  const float* eW1 = (const float*)d_in[7];   const float* eb1 = (const float*)d_in[8];
  const float* eW2 = (const float*)d_in[9];   const float* eb2 = (const float*)d_in[10];
  const float* cW1 = (const float*)d_in[11];  const float* cb1 = (const float*)d_in[12];
  const float* cW2 = (const float*)d_in[13];  const float* cb2 = (const float*)d_in[14];
  const float* nW1 = (const float*)d_in[15];  const float* nb1 = (const float*)d_in[16];
  const float* nW2 = (const float*)d_in[17];  const float* nb2 = (const float*)d_in[18];
  const float* mW1 = (const float*)d_in[19];  const float* mb1 = (const float*)d_in[20];
  const float* mW2 = (const float*)d_in[21];  const float* mb2 = (const float*)d_in[22];
  const float* oW1 = (const float*)d_in[23];  const float* ob1 = (const float*)d_in[24];
  const float* oW2 = (const float*)d_in[25];  const float* ob2 = (const float*)d_in[26];
  float* out = (float*)d_out;

  // ---- workspace carve ----
  char* ws = (char*)d_ws;
  size_t off = 0;
  auto carve = [&](size_t bytes) -> char* {
    char* p = ws + off;
    off += (bytes + 255) & ~(size_t)255;
    return p;
  };
  float* X0  = (float*)carve((size_t)NN * 42 * 4);
  float* X1  = (float*)carve((size_t)NN * 42 * 4);
  float* H   = (float*)carve((size_t)NN * HDIM * 4);
  float* agg = (float*)carve((size_t)NN * HDIM * 4);
  int*   nbr = (int*)carve((size_t)NN * KNB * 4);
  const size_t FR_EW1 = 9 * 8 * 512;   // fragment elems per layer (K=288,N=128)
  const size_t FR_128 = 4 * 8 * 512;   // K=128,N=128
  const size_t FR_NW1 = 8 * 8 * 512;   // K=256,N=128
  const size_t FR_MW2 = 4 * 4 * 512;   // K=128,N=64
  const size_t FR_WIN = 2 * 8 * 512;   // K=64, N=128
  const size_t FR_OW2 = 4 * 2 * 512;   // K=128,N=32
  _Float16* eW1f = (_Float16*)carve(3 * FR_EW1 * 2);
  _Float16* eW2f = (_Float16*)carve(3 * FR_128 * 2);
  _Float16* cW1f = (_Float16*)carve(3 * FR_128 * 2);
  _Float16* nW1f = (_Float16*)carve(3 * FR_NW1 * 2);
  _Float16* nW2f = (_Float16*)carve(3 * FR_128 * 2);
  _Float16* mW1f = (_Float16*)carve(FR_128 * 2);
  _Float16* mW2f = (_Float16*)carve(FR_MW2 * 2);
  _Float16* Winf = (_Float16*)carve(FR_WIN * 2);
  _Float16* oW1f = (_Float16*)carve(FR_128 * 2);
  _Float16* oW2f = (_Float16*)carve(FR_OW2 * 2);

  // ---- weight swizzle (f32 -> f16 fragment-major, zero-padded) ----
  auto swz = [&](_Float16* dst, const float* src, int Ktot, int ncols, int krows, int ncsrc) {
    int total = (Ktot >> 5) * (ncols >> 4) * 512;
    swz_kernel<<<(total + 255) / 256, 256, 0, stream>>>(dst, src, Ktot, ncols, krows, ncsrc);
  };
  for (int l = 0; l < NL; ++l) {
    swz(eW1f + l * FR_EW1, eW1 + (size_t)l * 257 * 128, 288, 128, 257, 128);
    swz(eW2f + l * FR_128, eW2 + (size_t)l * 128 * 128, 128, 128, 128, 128);
    swz(cW1f + l * FR_128, cW1 + (size_t)l * 128 * 128, 128, 128, 128, 128);
    swz(nW1f + l * FR_NW1, nW1 + (size_t)l * 256 * 128, 256, 128, 256, 128);
    swz(nW2f + l * FR_128, nW2 + (size_t)l * 128 * 128, 128, 128, 128, 128);
  }
  swz(mW1f, mW1, 128, 128, 128, 128);
  swz(mW2f, mW2, 128, 64, 128, 64);
  swz(Winf, W_in, 64, 128, 64, 128);
  swz(oW1f, oW1, 128, 128, 128, 128);
  swz(oW2f, oW2, 128, 32, 128, 25);

  // ---- copy input X to double buffer 0 (never mutate d_in) ----
  hipMemcpyAsync(X0, X_in, (size_t)NN * 42 * 4, hipMemcpyDeviceToDevice, stream);

  const size_t smem_mlp  = 4 * 16 * HDIM * sizeof(_Float16);                 // 16 KB
  const size_t smem_emb  = smem_mlp + 4 * 16 * ED * sizeof(_Float16);        // 24 KB
  const size_t smem_edge = (144 * 42 + 144 + 144 * HDIM) * sizeof(float)
                         + 9 * 16 * HDIM * sizeof(_Float16);                 // ~132 KB

  for (int t = 0; t < NROUND; ++t) {
    embed_kernel<<<NN / 64, 128, smem_emb, stream>>>(
        H, S, embed, Winf, b_in, mW1f, mb1, mW2f, mb2, (t > 0) ? 1 : 0);
    float* Xc0 = ((t * NL) & 1) ? X1 : X0;
    knn_kernel<<<NN / SEG, SEG, 0, stream>>>(Xc0, nbr);
    for (int l = 0; l < NL; ++l) {
      int lc = t * NL + l;
      float* Xc = (lc & 1) ? X1 : X0;
      float* Xn = (lc & 1) ? X0 : X1;
      edge_kernel<<<NN / 16, 288, smem_edge, stream>>>(
          Xc, Xn, H, agg, nbr, channel_w,
          eW1f + l * FR_EW1, eb1 + l * HDIM,
          eW2f + l * FR_128, eb2 + l * HDIM,
          cW1f + l * FR_128, cb1 + l * HDIM,
          cW2 + l * HDIM, cb2 + l);
      node_kernel<<<NN / 64, 128, smem_mlp, stream>>>(
          H, agg, nW1f + l * FR_NW1, nb1 + l * HDIM,
          nW2f + l * FR_128, nb2 + l * HDIM);
    }
  }

  // outputs: logits (N x 25) then X (N x 42); after 9 layers final X is in X1
  logits_kernel<<<NN / 64, 128, smem_mlp, stream>>>(H, oW1f, ob1, oW2f, ob2, out);
  copy_f32_kernel<<<(NN * 42 + 255) / 256, 256, 0, stream>>>(out + (size_t)NN * NCLS,
                                                             X1, NN * 42);
}
